// FlaxGemma2Attention_29652454212483
// MI455X (gfx1250) — matrix operand order repaired and checked by reading, not compiled, not run on hardware
//
#include <hip/hip_runtime.h>
#include <hip/hip_bf16.h>
#include <stdint.h>

// Problem constants (from reference)
#define B_      2
#define S_      2048
#define HID_    3584
#define NH_     16
#define NKV_    8
#define HD_     256
#define WINDOW_ 1024
#define SOFTCAP_ 50.0f

typedef __attribute__((ext_vector_type(16))) __bf16 v16bf;
typedef __attribute__((ext_vector_type(8)))  float  v8f;

union FragU { v16bf v; uint32_t u[8]; };

__device__ __forceinline__ uint16_t f2bf(float x) {
  union { float f; uint32_t u; } c; c.f = x;
  uint32_t r = (c.u + 0x7FFFu + ((c.u >> 16) & 1u)) >> 16;  // RNE
  return (uint16_t)r;
}

// ---- gfx1250 async global->LDS copy (ASYNCcnt-tracked data mover) ----------
#if defined(__has_builtin)
#if __has_builtin(__builtin_amdgcn_global_load_async_to_lds_b128)
#define HAVE_ASYNC_LDS_BUILTIN 1
#endif
#if __has_builtin(__builtin_amdgcn_s_wait_asynccnt)
#define HAVE_WAIT_ASYNC_BUILTIN 1
#endif
#endif

typedef int v4i_gcc __attribute__((vector_size(16)));   // matches builtin param type
typedef __attribute__((address_space(1))) v4i_gcc* glob_v4i;
typedef __attribute__((address_space(3))) v4i_gcc* lds_v4i;

__device__ __forceinline__ void async_copy_b128(const uint16_t* g, uint16_t* l) {
#if defined(HAVE_ASYNC_LDS_BUILTIN)
  __builtin_amdgcn_global_load_async_to_lds_b128(
      (glob_v4i)(uintptr_t)g,
      (lds_v4i)(uint32_t)(uintptr_t)l,
      0, 0);
#else
  asm volatile("global_load_async_to_lds_b128 %0, %1, off"
               :: "v"((uint32_t)(uintptr_t)l), "v"((uint64_t)(uintptr_t)g)
               : "memory");
#endif
}

__device__ __forceinline__ void wait_async_lds() {
#if defined(HAVE_WAIT_ASYNC_BUILTIN)
  __builtin_amdgcn_s_wait_asynccnt(0);
#else
  asm volatile("s_wait_asynccnt 0x0" ::: "memory");
#endif
}

// ---------------------------------------------------------------- convert
__global__ void k_cvt_bf16(const float* __restrict__ src,
                           uint16_t* __restrict__ dst, long n) {
  long i = (long)blockIdx.x * blockDim.x + threadIdx.x;
  long stride = (long)gridDim.x * blockDim.x;
  for (; i < n; i += stride) dst[i] = f2bf(src[i]);
}

// ---------------------------------------------------------------- GEMM
// C[M,N] (f32) = A[M,K] (bf16, row-major) @ B[K,N] (bf16, row-major)
// block: 256 threads = 8 waves; tile 128(M) x 128(N); k-step 32.
// A tile staged via async global->LDS (no transpose needed); B tile staged
// transposed (coalesced global b128 reads, scattered ds stores).
__global__ __launch_bounds__(256) void k_gemm(const uint16_t* __restrict__ A,
                                              const uint16_t* __restrict__ Bm,
                                              float* __restrict__ C,
                                              int M, int N, int K) {
  __shared__ __align__(16) uint16_t As[128 * 32];   // A tile, row-major stride 32
  __shared__ __align__(16) uint16_t Bt[128 * 40];   // B^T tile: Bt[n][k], padded
  const int lane = threadIdx.x & 31;
  const int wave = threadIdx.x >> 5;
  const int wr = wave >> 1, wc = wave & 1;          // 4x2 wave grid
  const int bn = blockIdx.x * 128;
  const int bm = blockIdx.y * 128;
  const int lh = lane >> 4, ll = lane & 15;

  v8f acc[2][4] = {};
  const int arow = threadIdx.x >> 1;                // 0..127 (A staging)
  const int aqh  = (threadIdx.x & 1) * 16;          // 0/16
  const int kloc = threadIdx.x >> 3;                // 0..31  (B staging)
  const int ngrp = threadIdx.x & 7;                 // 0..7 -> 16 n each

  for (int k0 = 0; k0 < K; k0 += 32) {
    __syncthreads();
    // ---- A tile: async DMA into LDS (2 x b128 per thread = 8KB total) ----
    {
      const uint16_t* gA = A + (size_t)(bm + arow) * K + k0 + aqh;
      async_copy_b128(gA,     &As[arow * 32 + aqh]);
      async_copy_b128(gA + 8, &As[arow * 32 + aqh + 8]);
    }
    // ---- B tile: coalesced global reads, transposed ds stores ----
    {
      const uint16_t* gB = Bm + (size_t)(k0 + kloc) * N + bn + ngrp * 16;
      uint4 p0 = *(const uint4*)gB;
      uint4 p1 = *(const uint4*)(gB + 8);
      const uint16_t* h0 = (const uint16_t*)&p0;
      const uint16_t* h1 = (const uint16_t*)&p1;
#pragma unroll
      for (int j = 0; j < 8; ++j) Bt[(ngrp * 16 + j) * 40 + kloc] = h0[j];
#pragma unroll
      for (int j = 0; j < 8; ++j) Bt[(ngrp * 16 + 8 + j) * 40 + kloc] = h1[j];
    }
    wait_async_lds();
    __syncthreads();

    FragU a[2], b[4];
#pragma unroll
    for (int i = 0; i < 2; ++i) {
      const int r = (wr * 32 + i * 16 + ll) * 32;
      *(uint4*)&a[i].u[0] = *(const uint4*)&As[r + lh * 8];
      *(uint4*)&a[i].u[4] = *(const uint4*)&As[r + 16 + lh * 8];
    }
#pragma unroll
    for (int j = 0; j < 4; ++j)
#pragma unroll
      for (int i = 0; i < 8; ++i)
        b[j].u[i] = *(const uint32_t*)
            &Bt[(wc * 64 + j * 16 + ll) * 40 + ((i >> 2) << 4) + (lh << 3) + ((i & 3) << 1)];

#pragma unroll
    for (int i = 0; i < 2; ++i)
#pragma unroll
      for (int j = 0; j < 4; ++j)
        acc[i][j] = __builtin_amdgcn_wmma_f32_16x16x32_bf16(
            false, a[i].v, false, b[j].v, (short)0, acc[i][j], false, false);
  }

#pragma unroll
  for (int i = 0; i < 2; ++i)
#pragma unroll
    for (int j = 0; j < 4; ++j)
#pragma unroll
      for (int r = 0; r < 8; ++r) {
        int row = bm + wr * 32 + i * 16 + lh * 8 + r;
        int col = bn + wc * 64 + j * 16 + ll;
        C[(size_t)row * N + col] = acc[i][j][r];
      }
}

// ---------------------------------------------------------------- RoPE
// x: [B][S][nh][HD] f32 -> out bf16 same layout; sin/cos: [S][HD]
__global__ void k_rope(const float* __restrict__ x, uint16_t* __restrict__ out,
                       const float* __restrict__ sinT, const float* __restrict__ cosT,
                       int nheads) {
  long n = (long)B_ * S_ * nheads * (HD_ / 2);
  long i = (long)blockIdx.x * blockDim.x + threadIdx.x;
  if (i >= n) return;
  int d = (int)(i % (HD_ / 2));
  long t = i / (HD_ / 2);
  int h = (int)(t % nheads);
  long bs = t / nheads;                       // b*S + s
  int s = (int)(bs % S_);
  size_t base = (size_t)bs * nheads * HD_ + (size_t)h * HD_;
  float x1 = x[base + d];
  float x2 = x[base + d + HD_ / 2];
  float c1 = cosT[(size_t)s * HD_ + d];
  float s1 = sinT[(size_t)s * HD_ + d];
  float c2 = cosT[(size_t)s * HD_ + d + HD_ / 2];
  float s2 = sinT[(size_t)s * HD_ + d + HD_ / 2];
  out[base + d]           = f2bf(x1 * c1 - x2 * s1);  // x*cos + rot_half(x)*sin
  out[base + d + HD_ / 2] = f2bf(x2 * c2 + x1 * s2);
}

// ---------------------------------------------------------------- attention
// q: [B][S][NH][HD] bf16 ; k,v: [B][S][NKV][HD] bf16 ; o: [B][S][NH][HD] bf16
// block = 128 threads (4 waves), each wave owns 16 query rows; 64 rows/block.
__global__ __launch_bounds__(128) void k_attn(const uint16_t* __restrict__ q,
                                              const uint16_t* __restrict__ kkg,
                                              const uint16_t* __restrict__ vvg,
                                              uint16_t* __restrict__ o) {
  __shared__ __align__(16) uint16_t Vt[256 * 40];   // V^T tile: Vt[d][sloc]
  __shared__ __align__(16) uint16_t Ps[4][16 * 40]; // per-wave P tile (C->A relayout)

  const int lane = threadIdx.x & 31;
  const int wave = threadIdx.x >> 5;
  const int lh = lane >> 4, ll = lane & 15;

  const int qt = blockIdx.x % (S_ / 64);
  const int h  = (blockIdx.x / (S_ / 64)) % NH_;
  const int b  = blockIdx.x / ((S_ / 64) * NH_);
  const int hk = h >> 1;                      // GQA: NH/NKV = 2
  const int q0 = qt * 64;
  const int qr0 = q0 + wave * 16;

  // resident Q A-fragments: 8 d-steps of 32
  FragU qf[8];
#pragma unroll
  for (int ds = 0; ds < 8; ++ds) {
#pragma unroll
    for (int i = 0; i < 8; ++i) {
      int koff = ((i >> 2) << 4) + (lh << 3) + ((i & 3) << 1);
      size_t idx = (((size_t)b * S_ + (qr0 + ll)) * NH_ + h) * HD_ + ds * 32 + koff;
      qf[ds].u[i] = *(const uint32_t*)&q[idx];
    }
  }

  float mrow[8], lrow[8];
#pragma unroll
  for (int r = 0; r < 8; ++r) { mrow[r] = -3.0e38f; lrow[r] = 0.0f; }
  v8f oacc[16] = {};

  int klo = q0 - (WINDOW_ - 1);
  if (klo < 0) klo = 0;
  klo &= ~31;
  const int khi = q0 + 64;
  const float scale = 0.0625f;                // HD^-0.5

  for (int k0 = klo; k0 < khi; k0 += 32) {
    // ---- cooperative V^T staging + prefetch of next K/V tiles ----
    __syncthreads();
    {
      int sloc = threadIdx.x >> 2;            // 0..31
      int dblk = (threadIdx.x & 3) * 64;
      const uint16_t* vrow =
          vvg + (((size_t)b * S_ + (k0 + sloc)) * NKV_ + hk) * HD_ + dblk;
      if (k0 + 32 < khi) {
        __builtin_prefetch(vrow + (size_t)32 * NKV_ * HD_, 0, 1);
        __builtin_prefetch(kkg + (((size_t)b * S_ + (k0 + 32 + sloc)) * NKV_ + hk) * HD_ + dblk, 0, 1);
      }
#pragma unroll
      for (int j = 0; j < 8; ++j) {
        uint4 pkt = *(const uint4*)(vrow + j * 8);
        const uint16_t* ph = (const uint16_t*)&pkt;
#pragma unroll
        for (int e = 0; e < 8; ++e)
          Vt[(dblk + j * 8 + e) * 40 + sloc] = ph[e];
      }
    }
    __syncthreads();

    if (k0 <= qr0 + 15) {                     // wave-uniform: shuffles are safe
      // ---- logits = Q @ K^T (16 x 32) ----
      v8f c0 = {}, c1 = {};
#pragma unroll
      for (int ds = 0; ds < 8; ++ds) {
        FragU kb0, kb1;
#pragma unroll
        for (int i = 0; i < 8; ++i) {
          int dk = ds * 32 + ((i >> 2) << 4) + (lh << 3) + ((i & 3) << 1);
          kb0.u[i] = *(const uint32_t*)
              &kkg[(((size_t)b * S_ + (k0 + ll)) * NKV_ + hk) * HD_ + dk];
          kb1.u[i] = *(const uint32_t*)
              &kkg[(((size_t)b * S_ + (k0 + 16 + ll)) * NKV_ + hk) * HD_ + dk];
        }
        c0 = __builtin_amdgcn_wmma_f32_16x16x32_bf16(false, qf[ds].v, false, kb0.v,
                                                     (short)0, c0, false, false);
        c1 = __builtin_amdgcn_wmma_f32_16x16x32_bf16(false, qf[ds].v, false, kb1.v,
                                                     (short)0, c1, false, false);
      }
      // ---- softcap + mask + online softmax ----
      float p0[8], p1[8];
#pragma unroll
      for (int r = 0; r < 8; ++r) {
        int qs = qr0 + lh * 8 + r;
        int ks0 = k0 + ll;
        int ks1 = k0 + 16 + ll;
        float x0 = SOFTCAP_ * tanhf(c0[r] * scale * (1.0f / SOFTCAP_));
        float x1 = SOFTCAP_ * tanhf(c1[r] * scale * (1.0f / SOFTCAP_));
        bool ok0 = (ks0 <= qs) && (ks0 > qs - WINDOW_);
        bool ok1 = (ks1 <= qs) && (ks1 > qs - WINDOW_);
        x0 = ok0 ? x0 : -3.0e38f;
        x1 = ok1 ? x1 : -3.0e38f;
        float rm = fmaxf(x0, x1);
        rm = fmaxf(rm, __shfl_xor(rm, 1, 32));
        rm = fmaxf(rm, __shfl_xor(rm, 2, 32));
        rm = fmaxf(rm, __shfl_xor(rm, 4, 32));
        rm = fmaxf(rm, __shfl_xor(rm, 8, 32));
        float mn = fmaxf(mrow[r], rm);
        float corr = __expf(mrow[r] - mn);
        float e0 = ok0 ? __expf(x0 - mn) : 0.0f;
        float e1 = ok1 ? __expf(x1 - mn) : 0.0f;
        float rs = e0 + e1;
        rs += __shfl_xor(rs, 1, 32);
        rs += __shfl_xor(rs, 2, 32);
        rs += __shfl_xor(rs, 4, 32);
        rs += __shfl_xor(rs, 8, 32);
        lrow[r] = lrow[r] * corr + rs;
        mrow[r] = mn;
        p0[r] = e0; p1[r] = e1;
#pragma unroll
        for (int dt = 0; dt < 16; ++dt) oacc[dt][r] *= corr;
      }
      // ---- P (C layout) -> bf16 A-fragment via per-wave LDS (in-order DS) ----
#pragma unroll
      for (int r = 0; r < 8; ++r) {
        int row = lh * 8 + r;
        Ps[wave][row * 40 + ll]      = f2bf(p0[r]);
        Ps[wave][row * 40 + 16 + ll] = f2bf(p1[r]);
      }
      FragU pf;
#pragma unroll
      for (int i = 0; i < 8; ++i) {
        int koff = ((i >> 2) << 4) + (lh << 3) + ((i & 3) << 1);
        pf.u[i] = *(const uint32_t*)&Ps[wave][ll * 40 + koff];
      }
      // ---- O += P @ V ----
#pragma unroll
      for (int dt = 0; dt < 16; ++dt) {
        FragU vb;
#pragma unroll
        for (int i = 0; i < 8; ++i)
          vb.u[i] = *(const uint32_t*)&Vt[(dt * 16 + ll) * 40 + ((i >> 2) << 4) + (lh << 3) + ((i & 3) << 1)];
        oacc[dt] = __builtin_amdgcn_wmma_f32_16x16x32_bf16(false, pf.v, false, vb.v,
                                                           (short)0, oacc[dt], false, false);
      }
    }
  }

  // ---- normalize + store bf16 [b][s][h][d] ----
#pragma unroll
  for (int r = 0; r < 8; ++r) lrow[r] = 1.0f / lrow[r];
#pragma unroll
  for (int dt = 0; dt < 16; ++dt)
#pragma unroll
    for (int r = 0; r < 8; ++r) {
      int s = qr0 + lh * 8 + r;
      int d = dt * 16 + ll;
      o[(((size_t)b * S_ + s) * NH_ + h) * HD_ + d] = f2bf(oacc[dt][r] * lrow[r]);
    }
}

// ---------------------------------------------------------------- launch
extern "C" void kernel_launch(void* const* d_in, const int* in_sizes, int n_in,
                              void* d_out, int out_size, void* d_ws, size_t ws_size,
                              hipStream_t stream) {
  const float* hidden = (const float*)d_in[0];
  const float* sinT   = (const float*)d_in[3];
  const float* cosT   = (const float*)d_in[4];
  const float* Wq     = (const float*)d_in[6];
  const float* Wk     = (const float*)d_in[7];
  const float* Wv     = (const float*)d_in[8];
  const float* Wo     = (const float*)d_in[9];
  float* out = (float*)d_out;

  const int M  = B_ * S_;          // 4096
  const int NQ = NH_ * HD_;        // 4096
  const int NK = NKV_ * HD_;       // 2048

  char* ws = (char*)d_ws;
  size_t off = 0;
  auto alloc = [&](size_t bytes) -> void* {
    void* p = ws + off;
    off = (off + bytes + 255) & ~(size_t)255;
    return p;
  };
  uint16_t* Xb  = (uint16_t*)alloc((size_t)M * HID_ * 2);
  uint16_t* Wqb = (uint16_t*)alloc((size_t)HID_ * NQ * 2);
  uint16_t* Wkb = (uint16_t*)alloc((size_t)HID_ * NK * 2);
  uint16_t* Wvb = (uint16_t*)alloc((size_t)HID_ * NK * 2);
  uint16_t* Wob = (uint16_t*)alloc((size_t)NQ * HID_ * 2);
  float*    qf  = (float*)   alloc((size_t)M * NQ * 4);
  float*    kf  = (float*)   alloc((size_t)M * NK * 4);
  float*    vf  = (float*)   alloc((size_t)M * NK * 4);
  uint16_t* qb  = (uint16_t*)alloc((size_t)M * NQ * 2);
  uint16_t* kb  = (uint16_t*)alloc((size_t)M * NK * 2);
  uint16_t* vb  = (uint16_t*)alloc((size_t)M * NK * 2);
  uint16_t* ab  = (uint16_t*)alloc((size_t)M * NQ * 2);

  // 1) bf16 conversions (activations + weights)
  k_cvt_bf16<<<4096, 256, 0, stream>>>(hidden, Xb, (long)M * HID_);
  k_cvt_bf16<<<4096, 256, 0, stream>>>(Wq, Wqb, (long)HID_ * NQ);
  k_cvt_bf16<<<4096, 256, 0, stream>>>(Wk, Wkb, (long)HID_ * NK);
  k_cvt_bf16<<<4096, 256, 0, stream>>>(Wv, Wvb, (long)HID_ * NK);
  k_cvt_bf16<<<4096, 256, 0, stream>>>(Wo, Wob, (long)NQ * HID_);

  // 2) QKV projections (WMMA GEMM, async-LDS A staging)
  k_gemm<<<dim3(NQ / 128, M / 128), 256, 0, stream>>>(Xb, Wqb, qf, M, NQ, HID_);
  k_gemm<<<dim3(NK / 128, M / 128), 256, 0, stream>>>(Xb, Wkb, kf, M, NK, HID_);
  k_gemm<<<dim3(NK / 128, M / 128), 256, 0, stream>>>(Xb, Wvb, vf, M, NK, HID_);

  // 3) RoPE (q, k) + bf16; v just converted
  {
    long nq = (long)B_ * S_ * NH_ * (HD_ / 2);
    long nk = (long)B_ * S_ * NKV_ * (HD_ / 2);
    k_rope<<<(unsigned)((nq + 255) / 256), 256, 0, stream>>>(qf, qb, sinT, cosT, NH_);
    k_rope<<<(unsigned)((nk + 255) / 256), 256, 0, stream>>>(kf, kb, sinT, cosT, NKV_);
    k_cvt_bf16<<<4096, 256, 0, stream>>>(vf, vb, (long)M * NK);
  }

  // 4) windowed flash attention (WMMA QK^T and PV)
  k_attn<<<B_ * NH_ * (S_ / 64), 128, 0, stream>>>(qb, kb, vb, ab);

  // 5) output projection -> d_out (f32)
  k_gemm<<<dim3(HID_ / 128, M / 128), 256, 0, stream>>>(ab, Wob, out, M, HID_, NQ);
}
